// MutualAttention_50457275793564
// MI455X (gfx1250) — matrix operand order repaired and checked by reading, not compiled, not run on hardware
//
#include <hip/hip_runtime.h>

// ---------------------------------------------------------------------------
// MutualAttention for MI455X (gfx1250): bf16 WMMA, wave32, 320KB WGP LDS
//   B=8, N=2048, D=512
//   out[b, 0:N]  = attn(q, k, q) ; out[b, N:2N] = attn(k, q, k)
// Kernel 2 processes 32 rows/workgroup: K and V^T fragments are each reused
// for two row-tiles, halving L2 traffic vs a 16-row design.
// ---------------------------------------------------------------------------

#define BATCH 8
#define NSEQ  2048
#define DMODEL 512
#define SCALE  0.044194173824159216f   // 1/sqrt(512)

#define QPITCH 520                     // Q-stage row pitch (bf16): banks spread
#define SPITCH 2056                    // S row pitch (bf16):       banks spread

typedef __attribute__((ext_vector_type(16))) __bf16 bf16x16;
typedef __attribute__((ext_vector_type(8)))  __bf16 bf16x8;
typedef __attribute__((ext_vector_type(8)))  float  f32x8;
typedef __attribute__((ext_vector_type(4)))  float  f32x4;
typedef __attribute__((ext_vector_type(4)))  unsigned int u32x4;

__device__ __forceinline__ __bf16 f2bf(float f) { return (__bf16)f; }
__device__ __forceinline__ unsigned short f2bf_bits(float f) {
    return __builtin_bit_cast(unsigned short, (__bf16)f);
}
__device__ __forceinline__ float bf2f(unsigned short h) {
    unsigned u = ((unsigned)h) << 16;
    return __builtin_bit_cast(float, u);
}

// ---- A-matrix fragment (16x32 bf16, MxK): lane%16 = row M,
//      per-lane K = (lane/16)*8 + {0..7} and +{16..23}  (ISA 7.12.2 table)
__device__ __forceinline__ bf16x16 frag_a_bf16(const unsigned short* base,
                                               int stride, int lane) {
    int row = lane & 15;
    int cb  = (lane >> 4) << 3;
    const unsigned short* p = base + row * stride + cb;
    u32x4 lo = *(const u32x4*)(p);
    u32x4 hi = *(const u32x4*)(p + 16);
    bf16x8 l = __builtin_bit_cast(bf16x8, lo);
    bf16x8 h = __builtin_bit_cast(bf16x8, hi);
    return __builtin_shufflevector(l, h, 0,1,2,3,4,5,6,7,8,9,10,11,12,13,14,15);
}

__device__ __forceinline__ bf16x16 frag_a_f32(const float* base,
                                              int stride, int lane) {
    int row = lane & 15;
    int cb  = (lane >> 4) << 3;
    const float* p = base + row * stride + cb;
    f32x4 a0 = *(const f32x4*)(p);
    f32x4 a1 = *(const f32x4*)(p + 4);
    f32x4 b0 = *(const f32x4*)(p + 16);
    f32x4 b1 = *(const f32x4*)(p + 20);
    bf16x16 r;
    r[0]=f2bf(a0[0]);  r[1]=f2bf(a0[1]);  r[2]=f2bf(a0[2]);  r[3]=f2bf(a0[3]);
    r[4]=f2bf(a1[0]);  r[5]=f2bf(a1[1]);  r[6]=f2bf(a1[2]);  r[7]=f2bf(a1[3]);
    r[8]=f2bf(b0[0]);  r[9]=f2bf(b0[1]);  r[10]=f2bf(b0[2]); r[11]=f2bf(b0[3]);
    r[12]=f2bf(b1[0]); r[13]=f2bf(b1[1]); r[14]=f2bf(b1[2]); r[15]=f2bf(b1[3]);
    return r;
}

// ---- B-matrix fragment (32x16 bf16, KxN): lane%16 = column N,
//      per-lane K = (lane/16)*8 + {0..7} and +{16..23}  (matches A table order)
__device__ __forceinline__ bf16x16 frag_b_bf16(const unsigned short* base,
                                               int stride, int lane) {
    int row = lane & 15;
    int cb  = (lane >> 4) << 3;
    const unsigned short* p = base + row * stride + cb;
    u32x4 lo = *(const u32x4*)(p);
    u32x4 hi = *(const u32x4*)(p + 16);
    bf16x8 l = __builtin_bit_cast(bf16x8, lo);
    bf16x8 h = __builtin_bit_cast(bf16x8, hi);
    return __builtin_shufflevector(l, h, 0,1,2,3,4,5,6,7,8,9,10,11,12,13,14,15);
}

__device__ __forceinline__ bf16x16 frag_b_f32(const float* base,
                                              int stride, int lane) {
    int row = lane & 15;
    int cb  = (lane >> 4) << 3;
    const float* p = base + row * stride + cb;
    f32x4 a0 = *(const f32x4*)(p);
    f32x4 a1 = *(const f32x4*)(p + 4);
    f32x4 b0 = *(const f32x4*)(p + 16);
    f32x4 b1 = *(const f32x4*)(p + 20);
    bf16x16 r;
    r[0]=f2bf(a0[0]);  r[1]=f2bf(a0[1]);  r[2]=f2bf(a0[2]);  r[3]=f2bf(a0[3]);
    r[4]=f2bf(a1[0]);  r[5]=f2bf(a1[1]);  r[6]=f2bf(a1[2]);  r[7]=f2bf(a1[3]);
    r[8]=f2bf(b0[0]);  r[9]=f2bf(b0[1]);  r[10]=f2bf(b0[2]); r[11]=f2bf(b0[3]);
    r[12]=f2bf(b1[0]); r[13]=f2bf(b1[1]); r[14]=f2bf(b1[2]); r[15]=f2bf(b1[3]);
    return r;
}

// ---------------------------------------------------------------------------
// Kernel 1: q = query@W1^T + b1 ; k = key@W2^T + b2 (bf16 + transposed copies)
// ---------------------------------------------------------------------------
__global__ __launch_bounds__(256)
void qk_linear_kernel(const float* __restrict__ query,
                      const float* __restrict__ key,
                      const float* __restrict__ W1, const float* __restrict__ b1,
                      const float* __restrict__ W2, const float* __restrict__ b2,
                      unsigned short* __restrict__ qws, unsigned short* __restrict__ kws,
                      unsigned short* __restrict__ qT,  unsigned short* __restrict__ kT)
{
    int tid  = threadIdx.x;
    int lane = tid & 31;
    int w    = blockIdx.x * 8 + (tid >> 5);   // 0 .. 65535
    int which = w >> 15;
    int rem   = w & 0x7FFF;
    int mt = rem >> 5;                        // row tile over B*N
    int et = rem & 31;                        // col tile over D
    int m0 = mt << 4, e0 = et << 4;

    const float* X  = which ? key : query;
    const float* Wm = which ? W2  : W1;
    const float* Bv = which ? b2  : b1;
    unsigned short* O  = which ? kws : qws;
    unsigned short* OT = which ? kT  : qT;

    f32x8 acc = {0.f,0.f,0.f,0.f,0.f,0.f,0.f,0.f};
#pragma unroll
    for (int c = 0; c < 16; ++c) {
        bf16x16 a = frag_a_f32(X  + m0 * DMODEL + c * 32, DMODEL, lane);
        bf16x16 b = frag_b_f32(Wm + e0 * DMODEL + c * 32, DMODEL, lane);
        acc = __builtin_amdgcn_wmma_f32_16x16x32_bf16(false, a, false, b,
                                                      (short)0, acc, false, false);
    }

    int colE  = e0 + (lane & 15);
    int rbase = (lane >> 4) << 3;
    float bias = Bv[colE];
#pragma unroll
    for (int r = 0; r < 8; ++r) {
        int mrow = m0 + rbase + r;
        unsigned short h = f2bf_bits(acc[r] + bias);
        O[mrow * DMODEL + colE] = h;
        int bb = mrow >> 11;
        int n  = mrow & (NSEQ - 1);
        OT[bb * (DMODEL * NSEQ) + colE * NSEQ + n] = h;
    }
}

// ---------------------------------------------------------------------------
// Kernel 2: 32 rows / workgroup. 8 waves, 256 threads.
//  LDS: Qs (32x520 bf16) | S (32x2056 bf16) | red (2x256 f32)  ~= 163 KB
//  Phase 0: async-stage Q tile to LDS (global_load_async_to_lds_b128).
//  Phase 1: S = scale*Q K^T, each K-fragment drives 2 WMMAs (2 row tiles).
//  Phase 2: row softmax (8 threads/row), P kept bf16 in LDS.
//  Phase 3: O = P V, each V^T fragment drives 2 WMMAs (2 row tiles).
// ---------------------------------------------------------------------------
__global__ __launch_bounds__(256)
void attention_kernel(const unsigned short* __restrict__ qws,
                      const unsigned short* __restrict__ kws,
                      const unsigned short* __restrict__ qT,
                      const unsigned short* __restrict__ kT,
                      float* __restrict__ out)
{
    extern __shared__ unsigned short smem[];
    unsigned short* Qs = smem;                         // 32 x QPITCH
    unsigned short* S  = smem + 32 * QPITCH;           // 32 x SPITCH
    float* red  = (float*)(smem + 32 * QPITCH + 32 * SPITCH);
    float* red2 = red + 256;

    int tid  = threadIdx.x;
    int lane = tid & 31;
    int wave = tid >> 5;
    int mt    = blockIdx.x;             // 0..63  (32-row blocks)
    int b     = blockIdx.y;             // 0..7
    int which = blockIdx.z;             // 0: attn(q,k,q)  1: attn(k,q,k)

    const unsigned short* Q  = (which ? kws : qws) + (size_t)b * NSEQ * DMODEL;
    const unsigned short* K  = (which ? qws : kws) + (size_t)b * NSEQ * DMODEL;
    const unsigned short* VT = (which ? kT  : qT ) + (size_t)b * DMODEL * NSEQ;

    int m0    = mt << 5;                // first of 32 rows
    int colw  = lane & 15;
    int rbase = (lane >> 4) << 3;

    // ------- phase 0: stage Q[m0:m0+32][0:512] into LDS (async copy) -------
    {
        // 32 rows x 1KB ; 256 threads x 16B per pass x 8 passes
#pragma unroll
        for (int pass = 0; pass < 8; ++pass) {
            int row = (tid >> 6) + (pass << 2);          // 4 rows per pass
            int ch  = tid & 63;                           // 16B chunk in row
            const unsigned short* gp = Q + (size_t)(m0 + row) * DMODEL + (ch << 3);
            unsigned ldsoff = (unsigned)((row * QPITCH + (ch << 3)) * 2);
            asm volatile("global_load_async_to_lds_b128 %0, %1, off"
                         :: "v"(ldsoff), "v"(gp) : "memory");
        }
        asm volatile("s_wait_asynccnt 0" ::: "memory");
    }
    __syncthreads();

    // ------- phase 1: S = scale * Q K^T  (wave owns 256-column slice) ------
    int js = wave << 8;
    for (int jt = 0; jt < 16; ++jt) {
        int n0 = js + (jt << 4);
        f32x8 acc0 = {0.f,0.f,0.f,0.f,0.f,0.f,0.f,0.f};
        f32x8 acc1 = {0.f,0.f,0.f,0.f,0.f,0.f,0.f,0.f};
#pragma unroll
        for (int c = 0; c < 16; ++c) {
            bf16x16 kf = frag_b_bf16(K + n0 * DMODEL + c * 32, DMODEL, lane);
            bf16x16 q0 = frag_a_bf16(Qs + c * 32,               QPITCH, lane);
            bf16x16 q1 = frag_a_bf16(Qs + 16 * QPITCH + c * 32, QPITCH, lane);
            acc0 = __builtin_amdgcn_wmma_f32_16x16x32_bf16(false, q0, false, kf,
                                                           (short)0, acc0, false, false);
            acc1 = __builtin_amdgcn_wmma_f32_16x16x32_bf16(false, q1, false, kf,
                                                           (short)0, acc1, false, false);
        }
#pragma unroll
        for (int r = 0; r < 8; ++r) {
            S[(rbase + r) * SPITCH + n0 + colw]        = f2bf_bits(acc0[r] * SCALE);
            S[(16 + rbase + r) * SPITCH + n0 + colw]   = f2bf_bits(acc1[r] * SCALE);
        }
    }
    __syncthreads();

    // ------- phase 2: row softmax over 2048 cols (8 threads per row) -------
    {
        int row = tid >> 3, ts = tid & 7;
        unsigned short* sr = S + row * SPITCH;
        float m = -3.402823466e38f;
        for (int i = 0; i < 256; ++i)
            m = fmaxf(m, bf2f(sr[ts + (i << 3)]));
        red[row * 8 + ts] = m;
        __syncthreads();
        float mr = -3.402823466e38f;
#pragma unroll
        for (int kk = 0; kk < 8; ++kk) mr = fmaxf(mr, red[row * 8 + kk]);
        float s = 0.f;
        for (int i = 0; i < 256; ++i) {
            float e = __expf(bf2f(sr[ts + (i << 3)]) - mr);
            sr[ts + (i << 3)] = f2bf_bits(e);
            s += e;
        }
        red2[row * 8 + ts] = s;
        __syncthreads();
        float tot = 0.f;
#pragma unroll
        for (int kk = 0; kk < 8; ++kk) tot += red2[row * 8 + kk];
        float inv = 1.0f / tot;
        for (int i = 0; i < 256; ++i)
            sr[ts + (i << 3)] = f2bf_bits(bf2f(sr[ts + (i << 3)]) * inv);
    }
    __syncthreads();

    // ------- phase 3: O = P V  (wave owns 64 output columns) ---------------
    int d0 = wave << 6;
    f32x8 z = {0.f,0.f,0.f,0.f,0.f,0.f,0.f,0.f};
    f32x8 o[8] = { z, z, z, z, z, z, z, z };
    for (int j = 0; j < NSEQ; j += 32) {
        bf16x16 p0 = frag_a_bf16(S + j,               SPITCH, lane);
        bf16x16 p1 = frag_a_bf16(S + 16 * SPITCH + j, SPITCH, lane);
#pragma unroll
        for (int t = 0; t < 4; ++t) {
            bf16x16 vf = frag_b_bf16(VT + (size_t)(d0 + (t << 4)) * NSEQ + j,
                                     NSEQ, lane);
            o[t]     = __builtin_amdgcn_wmma_f32_16x16x32_bf16(false, p0, false, vf,
                                                               (short)0, o[t],     false, false);
            o[4 + t] = __builtin_amdgcn_wmma_f32_16x16x32_bf16(false, p1, false, vf,
                                                               (short)0, o[4 + t], false, false);
        }
    }

    float* ob = out + (size_t)b * (2 * NSEQ * DMODEL)
                    + (size_t)(which * NSEQ + m0) * DMODEL;
#pragma unroll
    for (int t = 0; t < 4; ++t)
#pragma unroll
        for (int r = 0; r < 8; ++r) {
            ob[(rbase + r) * DMODEL + d0 + (t << 4) + colw]        = o[t][r];
            ob[(16 + rbase + r) * DMODEL + d0 + (t << 4) + colw]   = o[4 + t][r];
        }
}

// ---------------------------------------------------------------------------
extern "C" void kernel_launch(void* const* d_in, const int* in_sizes, int n_in,
                              void* d_out, int out_size, void* d_ws, size_t ws_size,
                              hipStream_t stream)
{
    (void)in_sizes; (void)n_in; (void)out_size; (void)ws_size;
    const float* query = (const float*)d_in[0];
    const float* key   = (const float*)d_in[1];
    const float* W1    = (const float*)d_in[2];
    const float* b1    = (const float*)d_in[3];
    const float* W2    = (const float*)d_in[4];
    const float* b2    = (const float*)d_in[5];
    float* out = (float*)d_out;

    const size_t SZ = (size_t)BATCH * NSEQ * DMODEL;
    unsigned short* qws = (unsigned short*)d_ws;
    unsigned short* kws = qws + SZ;
    unsigned short* qT  = kws + SZ;
    unsigned short* kT  = qT  + SZ;

    qk_linear_kernel<<<8192, 256, 0, stream>>>(query, key, W1, b1, W2, b2,
                                               qws, kws, qT, kT);

    // LDS: 32*520*2 + 32*2056*2 + 2*256*4 = 33280 + 131584 + 2048 = 166912 B
    size_t lds = (size_t)(32 * QPITCH + 32 * SPITCH) * 2 + 2 * 256 * 4;
    hipFuncSetAttribute((const void*)attention_kernel,
                        hipFuncAttributeMaxDynamicSharedMemorySize, (int)lds);
    attention_kernel<<<dim3(64, BATCH, 2), 256, lds, stream>>>(qws, kws, qT, kT, out);
}
